// InvariantGNNBlock_3066606649476
// MI455X (gfx1250) — matrix operand order repaired and checked by reading, not compiled, not run on hardware
//
#include <hip/hip_runtime.h>
#include <hip/hip_bf16.h>
#include <stdint.h>

// ---------------------------------------------------------------------------
// CDNA5 (gfx1250) bf16 WMMA types / helpers
// ---------------------------------------------------------------------------
typedef __attribute__((ext_vector_type(16))) __bf16 bf16x16;
typedef __attribute__((ext_vector_type(8)))  float  f32x8;

union FragU {
    bf16x16        v;
    uint4          q[2];
    unsigned short s[16];
};

__device__ __forceinline__ unsigned short f2bf(float f) {
    union { float f; uint32_t u; } x; x.f = f;
    uint32_t r = x.u + 0x7FFFu + ((x.u >> 16) & 1u);   // round-to-nearest-even
    return (unsigned short)(r >> 16);
}

// A-matrix 16x32 bf16 fragment (one 16-edge/node tile row per lane).
// lane holds row (lane&15); k slots: [kb..kb+7] and [kb+16..kb+23], kb=k0+(lane>>4)*8
__device__ __forceinline__ bf16x16 load_a_frag(const unsigned short* rowptr, int k0, int lane) {
    int kb = k0 + ((lane >> 4) << 3);
    FragU f;
    f.q[0] = *(const uint4*)(rowptr + kb);
    f.q[1] = *(const uint4*)(rowptr + kb + 16);
    return f.v;
}

// B-matrix 32x16 bf16 fragment from transposed weights WT[Dout][K] (row-major).
// lane holds column n = ntile*16 + (lane&15); k slots: [kb..kb+7] and [kb+16..kb+23], kb=k0+(lane>>4)*8
__device__ __forceinline__ bf16x16 load_b_frag(const unsigned short* WT, int K,
                                               int ntile, int k0, int lane) {
    const unsigned short* p = WT + (size_t)(ntile * 16 + (lane & 15)) * K + k0 + ((lane >> 4) << 3);
    FragU f;
    f.q[0] = *(const uint4*)(p);
    f.q[1] = *(const uint4*)(p + 16);
    return f.v;
}

__device__ __forceinline__ f32x8 wmma_bf16(bf16x16 a, bf16x16 b, f32x8 c) {
    return __builtin_amdgcn_wmma_f32_16x16x32_bf16(false, a, false, b, (short)0, c, false, false);
}

// float atomic max via monotonic integer-bits trick (init must be -inf)
__device__ __forceinline__ void atomicMaxF(float* addr, float val) {
    if (val >= 0.f) atomicMax((int*)addr, __float_as_int(val));
    else            atomicMin((unsigned int*)addr, (unsigned int)__float_as_int(val));
}

#define LDS_STRIDE 136   // 128 + 8 bf16 pad -> 272B rows, rotates LDS banks

// Compute e-tile for 16 edges: e = relu(a*We1+be1) @ We2 + be2, result (bf16)
// written to LDS in row-major [16][LDS_STRIDE] so it can be re-read as A-frags.
// B stream is double-buffered in registers: load of frag f+1 issues before the
// WMMA consuming frag f, so VMEM latency overlaps the matrix pipe.
__device__ __forceinline__ void compute_e_tile(float a, const float* We1, const float* be1,
                                               const unsigned short* We2T, const float* be2,
                                               unsigned short* eL, int lane) {
    int m   = lane & 15;
    int kb0 = (lane >> 4) << 3;
    bf16x16 ef[4];
#pragma unroll
    for (int f = 0; f < 4; ++f) {
        FragU fr;
        int kk = f * 32 + kb0;
#pragma unroll
        for (int j = 0; j < 8; ++j) {
            float h0 = fmaxf(fmaf(a, We1[kk + j],      be1[kk + j]),      0.f);
            float h1 = fmaxf(fmaf(a, We1[kk + 16 + j], be1[kk + 16 + j]), 0.f);
            fr.s[j]     = f2bf(h0);
            fr.s[j + 8] = f2bf(h1);
        }
        ef[f] = fr.v;
    }
    int mb = (lane >> 4) << 3;
    bf16x16 bcur = load_b_frag(We2T, 128, 0, 0, lane);
    for (int nt = 0; nt < 8; ++nt) {
        float bias = be2[nt * 16 + m];
        f32x8 acc;
#pragma unroll
        for (int j = 0; j < 8; ++j) acc[j] = bias;
#pragma unroll
        for (int f = 0; f < 4; ++f) {
            bf16x16 bnext = (f < 3)
                ? load_b_frag(We2T, 128, nt, (f + 1) * 32, lane)
                : load_b_frag(We2T, 128, min(nt + 1, 7), 0, lane);
            acc = wmma_bf16(ef[f], bcur, acc);
            bcur = bnext;
        }
#pragma unroll
        for (int j = 0; j < 8; ++j) eL[(mb + j) * LDS_STRIDE + nt * 16 + m] = f2bf(acc[j]);
    }
    __builtin_amdgcn_wave_barrier();
}

// ---------------------------------------------------------------------------
// Kernel 0: init output (h copy + pos tail) and segment buffers
// ---------------------------------------------------------------------------
__global__ void init_kernel(const float* __restrict__ h, const float* __restrict__ pos,
                            float* __restrict__ out, float* __restrict__ segmax,
                            float* __restrict__ segsum, int N) {
    int i  = blockIdx.x * blockDim.x + threadIdx.x;
    int ND = N * 128;
    if (i < ND)    out[i] = h[i];
    if (i < N)   { segmax[i] = -__builtin_huge_valf(); segsum[i] = 0.f; }
    if (i < 2 * N) out[ND + i] = pos[i];
}

__global__ void cvt_bf16_kernel(const float* __restrict__ src,
                                unsigned short* __restrict__ dst, int n) {
    int i = blockIdx.x * blockDim.x + threadIdx.x;
    if (i < n) dst[i] = f2bf(src[i]);
}

// dst[Nout][K] (bf16) = transpose of src[K][Nout] (f32)
__global__ void transpose_bf16_kernel(const float* __restrict__ src,
                                      unsigned short* __restrict__ dst, int K, int Nout) {
    int i = blockIdx.x * blockDim.x + threadIdx.x;
    if (i >= K * Nout) return;
    int n = i / K, k = i - n * K;
    dst[i] = f2bf(src[k * Nout + n]);
}

// pack bq|bk|bv into one contiguous [3][128] buffer (offset math, no ptr selects)
__global__ void pack_bias_kernel(const float* __restrict__ bq, const float* __restrict__ bk,
                                 const float* __restrict__ bv, float* __restrict__ bqkv) {
    int i = blockIdx.x * blockDim.x + threadIdx.x;
    if (i < 128)      bqkv[i] = bq[i];
    else if (i < 256) bqkv[i] = bk[i - 128];
    else if (i < 384) bqkv[i] = bv[i - 256];
}

// ---------------------------------------------------------------------------
// Kernel 1: q/k/v = h @ W + b  (bf16 WMMA, bf16 outputs)
// One 16-row m-tile per wave; the wave walks all 24 (matrix, ntile) tasks so
// its A-fragments are loaded exactly once. Weights/biases/outputs are packed
// contiguously (offset math only). B stream double-buffered across tasks.
// ---------------------------------------------------------------------------
__global__ void qkv_kernel(const unsigned short* __restrict__ hb,
                           const unsigned short* __restrict__ WqkvT,  // [3][128][128]
                           const float* __restrict__ bqkv,            // [3][128]
                           unsigned short* __restrict__ qkvb,         // [3][N][128]
                           int N, int mtiles) {
    int wave = threadIdx.x >> 5, lane = threadIdx.x & 31;
    int mtile = blockIdx.x * 8 + wave;
    if (mtile >= mtiles) return;
    int m = lane & 15;
    const unsigned short* hrow = hb + (size_t)(mtile * 16 + m) * 128;

    bf16x16 afr[4];
#pragma unroll
    for (int f = 0; f < 4; ++f) afr[f] = load_a_frag(hrow, f * 32, lane);

    int mb = (lane >> 4) << 3;
    bf16x16 bcur = load_b_frag(WqkvT, 128, 0, 0, lane);
    for (int task = 0; task < 24; ++task) {
        int which = task >> 3;        // 0=q, 1=k, 2=v
        int nt    = task & 7;
        const unsigned short* WT = WqkvT + (size_t)which * (128 * 128);
        float bias = bqkv[which * 128 + nt * 16 + m];
        f32x8 acc;
#pragma unroll
        for (int j = 0; j < 8; ++j) acc[j] = bias;
#pragma unroll
        for (int f = 0; f < 4; ++f) {
            bf16x16 bnext;
            if (f < 3) bnext = load_b_frag(WT, 128, nt, (f + 1) * 32, lane);
            else {
                int tn = (task + 1 < 24) ? task + 1 : 23;
                bnext = load_b_frag(WqkvT + (size_t)(tn >> 3) * (128 * 128), 128, tn & 7, 0, lane);
            }
            acc = wmma_bf16(afr[f], bcur, acc);
            bcur = bnext;
        }
        size_t obase = ((size_t)which * N + (size_t)mtile * 16) * 128;
#pragma unroll
        for (int j = 0; j < 8; ++j)
            qkvb[obase + (size_t)(mb + j) * 128 + nt * 16 + m] = f2bf(acc[j]);
    }
}

// ---------------------------------------------------------------------------
// Kernel 2: per-edge attention logits + segment max
// one wave handles a 16-edge tile: attn_in = [q[col] | k[row] | e] (16 x 384)
// hidden = relu(attn_in @ Wa1 + ba1); logit = hidden . Wa2 + ba2
// ---------------------------------------------------------------------------
__global__ void logits_kernel(const unsigned short* __restrict__ qb,
                              const unsigned short* __restrict__ kbuf,
                              const long long* __restrict__ eidx,
                              const float* __restrict__ edge_attr,
                              const float* __restrict__ We1, const float* __restrict__ be1,
                              const unsigned short* __restrict__ We2T,
                              const float* __restrict__ be2,
                              const unsigned short* __restrict__ Wa1T,
                              const float* __restrict__ ba1,
                              const float* __restrict__ Wa2, const float* __restrict__ ba2,
                              float* __restrict__ logits, float* __restrict__ segmax,
                              int E, int ntiles) {
    __shared__ __align__(16) unsigned short lds_e[8][16 * LDS_STRIDE];
    int wave = threadIdx.x >> 5, lane = threadIdx.x & 31;
    int tile = blockIdx.x * 8 + wave;
    if (tile >= ntiles) return;
    int base = tile * 16;
    const long long* rowi = eidx;
    const long long* coli = eidx + E;
    int m = lane & 15;
    int eSelf = min(base + m, E - 1);
    int colm = (int)coli[eSelf];
    int rowm = (int)rowi[eSelf];

    // prefetch the gathered q/k rows (2 x 128B each); the e-tile WMMA work
    // below shadows the HBM latency of these random-access fetches.
    const unsigned short* qrow = qb   + (size_t)colm * 128;
    const unsigned short* krow = kbuf + (size_t)rowm * 128;
    __builtin_prefetch(qrow, 0, 3);
    __builtin_prefetch(qrow + 64, 0, 3);
    __builtin_prefetch(krow, 0, 3);
    __builtin_prefetch(krow + 64, 0, 3);

    unsigned short* eL = lds_e[wave];
    compute_e_tile(edge_attr[eSelf], We1, be1, We2T, be2, eL, lane);

    bf16x16 afr[12];
#pragma unroll
    for (int f = 0; f < 4; ++f) afr[f]     = load_a_frag(qrow, f * 32, lane);
#pragma unroll
    for (int f = 0; f < 4; ++f) afr[4 + f] = load_a_frag(krow, f * 32, lane);
#pragma unroll
    for (int f = 0; f < 4; ++f) afr[8 + f] = load_a_frag(eL + m * LDS_STRIDE, f * 32, lane);

    float p[8];
#pragma unroll
    for (int j = 0; j < 8; ++j) p[j] = 0.f;

    bf16x16 bcur = load_b_frag(Wa1T, 384, 0, 0, lane);
    for (int nt = 0; nt < 8; ++nt) {
        float bias = ba1[nt * 16 + m];
        f32x8 acc;
#pragma unroll
        for (int j = 0; j < 8; ++j) acc[j] = bias;
#pragma unroll
        for (int f = 0; f < 12; ++f) {
            bf16x16 bnext = (f < 11)
                ? load_b_frag(Wa1T, 384, nt, (f + 1) * 32, lane)
                : load_b_frag(Wa1T, 384, min(nt + 1, 7), 0, lane);
            acc = wmma_bf16(afr[f], bcur, acc);
            bcur = bnext;
        }
        float wa = Wa2[nt * 16 + m];
#pragma unroll
        for (int j = 0; j < 8; ++j) p[j] += fmaxf(acc[j], 0.f) * wa;
    }
    // sum over the 16 lanes that hold the same rows (columns n mod 16)
    for (int mask = 1; mask < 16; mask <<= 1) {
#pragma unroll
        for (int j = 0; j < 8; ++j) p[j] += __shfl_xor(p[j], mask, 32);
    }
    if (m == 0) {
        float bb = ba2[0];
        int mb = (lane >> 4) << 3;
        for (int j = 0; j < 8; ++j) {
            int eg = base + mb + j;
            if (eg < E) {
                float lg = p[j] + bb;
                logits[eg] = lg;
                atomicMaxF(&segmax[(int)coli[eg]], lg);
            }
        }
    }
}

// ---------------------------------------------------------------------------
// Kernel 3: ex = exp(logit - segmax[col]); segsum[col] += ex   (ex overwrites logits)
// ---------------------------------------------------------------------------
__global__ void ex_kernel(const long long* __restrict__ coli,
                          const float* __restrict__ segmax,
                          float* __restrict__ logits_ex,
                          float* __restrict__ segsum, int E) {
    int i = blockIdx.x * blockDim.x + threadIdx.x;
    if (i >= E) return;
    int c = (int)coli[i];
    float ex = __expf(logits_ex[i] - segmax[c]);
    logits_ex[i] = ex;
    atomicAdd(&segsum[c], ex);
}

// ---------------------------------------------------------------------------
// Kernel 4: messages + weighted scatter-add into out
// msg = silu([v[row] | e] @ Wm1 + bm1) @ Wm2 + bm2 ; out[col] += msg * w
// ---------------------------------------------------------------------------
__global__ void msg_kernel(const unsigned short* __restrict__ vb,
                           const long long* __restrict__ eidx,
                           const float* __restrict__ edge_attr,
                           const float* __restrict__ We1, const float* __restrict__ be1,
                           const unsigned short* __restrict__ We2T,
                           const float* __restrict__ be2,
                           const unsigned short* __restrict__ Wm1T,
                           const float* __restrict__ bm1,
                           const unsigned short* __restrict__ Wm2T,
                           const float* __restrict__ bm2,
                           const float* __restrict__ exbuf,
                           const float* __restrict__ segsum,
                           float* __restrict__ out, int E, int ntiles) {
    __shared__ __align__(16) unsigned short lds_t[8][16 * LDS_STRIDE];
    __shared__ float lds_w[8][16];
    __shared__ int   lds_c[8][16];
    int wave = threadIdx.x >> 5, lane = threadIdx.x & 31;
    int tile = blockIdx.x * 8 + wave;
    if (tile >= ntiles) return;
    int base = tile * 16;
    const long long* rowi = eidx;
    const long long* coli = eidx + E;
    int m = lane & 15;
    int eSelf = min(base + m, E - 1);
    int colm = (int)coli[eSelf];
    int rowm = (int)rowi[eSelf];

    const unsigned short* vrow = vb + (size_t)rowm * 128;
    __builtin_prefetch(vrow, 0, 3);
    __builtin_prefetch(vrow + 64, 0, 3);

    if (lane < 16) {
        float wv = 0.f;
        if (base + m < E) wv = exbuf[eSelf] / (segsum[colm] + 1e-16f);
        lds_w[wave][m] = wv;
        lds_c[wave][m] = colm;
    }

    unsigned short* tL = lds_t[wave];
    compute_e_tile(edge_attr[eSelf], We1, be1, We2T, be2, tL, lane);

    bf16x16 afr[8];
#pragma unroll
    for (int f = 0; f < 4; ++f) afr[f]     = load_a_frag(vrow, f * 32, lane);
#pragma unroll
    for (int f = 0; f < 4; ++f) afr[4 + f] = load_a_frag(tL + m * LDS_STRIDE, f * 32, lane);
    __builtin_amdgcn_wave_barrier();

    int mb = (lane >> 4) << 3;
    // hidden = silu(msg_in @ Wm1 + bm1) -> back to LDS for layout conversion
    {
        bf16x16 bcur = load_b_frag(Wm1T, 256, 0, 0, lane);
        for (int nt = 0; nt < 8; ++nt) {
            float bias = bm1[nt * 16 + m];
            f32x8 acc;
#pragma unroll
            for (int j = 0; j < 8; ++j) acc[j] = bias;
#pragma unroll
            for (int f = 0; f < 8; ++f) {
                bf16x16 bnext = (f < 7)
                    ? load_b_frag(Wm1T, 256, nt, (f + 1) * 32, lane)
                    : load_b_frag(Wm1T, 256, min(nt + 1, 7), 0, lane);
                acc = wmma_bf16(afr[f], bcur, acc);
                bcur = bnext;
            }
#pragma unroll
            for (int j = 0; j < 8; ++j) {
                float s = acc[j];
                float act = s / (1.f + __expf(-s));        // silu
                tL[(mb + j) * LDS_STRIDE + nt * 16 + m] = f2bf(act);
            }
        }
    }
    __builtin_amdgcn_wave_barrier();

    bf16x16 hfr[4];
#pragma unroll
    for (int f = 0; f < 4; ++f) hfr[f] = load_a_frag(tL + m * LDS_STRIDE, f * 32, lane);

    bf16x16 bcur = load_b_frag(Wm2T, 128, 0, 0, lane);
    for (int nt = 0; nt < 8; ++nt) {
        float bias = bm2[nt * 16 + m];
        f32x8 acc;
#pragma unroll
        for (int j = 0; j < 8; ++j) acc[j] = bias;
#pragma unroll
        for (int f = 0; f < 4; ++f) {
            bf16x16 bnext = (f < 3)
                ? load_b_frag(Wm2T, 128, nt, (f + 1) * 32, lane)
                : load_b_frag(Wm2T, 128, min(nt + 1, 7), 0, lane);
            acc = wmma_bf16(hfr[f], bcur, acc);
            bcur = bnext;
        }
#pragma unroll
        for (int j = 0; j < 8; ++j) {
            int   mm = mb + j;
            float wv = lds_w[wave][mm];
            int   c  = lds_c[wave][mm];
            atomicAdd(&out[(size_t)c * 128 + nt * 16 + m], acc[j] * wv);
        }
    }
}

// ---------------------------------------------------------------------------
// Host launch
// ---------------------------------------------------------------------------
extern "C" void kernel_launch(void* const* d_in, const int* in_sizes, int n_in,
                              void* d_out, int out_size, void* d_ws, size_t ws_size,
                              hipStream_t stream) {
    const float*     h    = (const float*)d_in[0];
    const float*     pos  = (const float*)d_in[1];
    const long long* eidx = (const long long*)d_in[2];
    const float*     ea   = (const float*)d_in[3];
    const float* Wq = (const float*)d_in[4];  const float* bq = (const float*)d_in[5];
    const float* Wk = (const float*)d_in[6];  const float* bk = (const float*)d_in[7];
    const float* Wv = (const float*)d_in[8];  const float* bv = (const float*)d_in[9];
    const float* We1 = (const float*)d_in[10]; const float* be1 = (const float*)d_in[11];
    const float* We2 = (const float*)d_in[12]; const float* be2 = (const float*)d_in[13];
    const float* Wa1 = (const float*)d_in[14]; const float* ba1 = (const float*)d_in[15];
    const float* Wa2 = (const float*)d_in[16]; const float* ba2 = (const float*)d_in[17];
    const float* Wm1 = (const float*)d_in[18]; const float* bm1 = (const float*)d_in[19];
    const float* Wm2 = (const float*)d_in[20]; const float* bm2 = (const float*)d_in[21];

    const int N = in_sizes[0] / 128;
    const int E = in_sizes[3];
    float* out = (float*)d_out;

    // workspace carve-up
    char* ws = (char*)d_ws;
    auto alloc = [&](size_t bytes) { char* p = ws; ws += (bytes + 255) & ~(size_t)255; return p; };
    unsigned short* hb    = (unsigned short*)alloc((size_t)N * 128 * 2);
    unsigned short* qkvb  = (unsigned short*)alloc((size_t)3 * N * 128 * 2);  // q|k|v packed
    unsigned short* WqkvT = (unsigned short*)alloc((size_t)3 * 128 * 128 * 2);
    unsigned short* We2T  = (unsigned short*)alloc(128 * 128 * 2);
    unsigned short* Wa1T  = (unsigned short*)alloc(128 * 384 * 2);
    unsigned short* Wm1T  = (unsigned short*)alloc(128 * 256 * 2);
    unsigned short* Wm2T  = (unsigned short*)alloc(128 * 128 * 2);
    float* bqkv   = (float*)alloc(3 * 128 * 4);
    float* logits = (float*)alloc((size_t)E * 4);
    float* segmax = (float*)alloc((size_t)N * 4);
    float* segsum = (float*)alloc((size_t)N * 4);

    const unsigned short* qb   = qkvb;
    const unsigned short* kbuf = qkvb + (size_t)N * 128;
    const unsigned short* vb   = qkvb + (size_t)2 * N * 128;

    const int B = 256;
    // 0) init
    init_kernel<<<(N * 128 + B - 1) / B, B, 0, stream>>>(h, pos, out, segmax, segsum, N);
    // 1) conversions / weight transposes / bias packing
    cvt_bf16_kernel<<<(N * 128 + B - 1) / B, B, 0, stream>>>(h, hb, N * 128);
    transpose_bf16_kernel<<<(128 * 128 + B - 1) / B, B, 0, stream>>>(Wq,  WqkvT,                 128, 128);
    transpose_bf16_kernel<<<(128 * 128 + B - 1) / B, B, 0, stream>>>(Wk,  WqkvT + 128 * 128,     128, 128);
    transpose_bf16_kernel<<<(128 * 128 + B - 1) / B, B, 0, stream>>>(Wv,  WqkvT + 2 * 128 * 128, 128, 128);
    transpose_bf16_kernel<<<(128 * 128 + B - 1) / B, B, 0, stream>>>(We2, We2T, 128, 128);
    transpose_bf16_kernel<<<(384 * 128 + B - 1) / B, B, 0, stream>>>(Wa1, Wa1T, 384, 128);
    transpose_bf16_kernel<<<(256 * 128 + B - 1) / B, B, 0, stream>>>(Wm1, Wm1T, 256, 128);
    transpose_bf16_kernel<<<(128 * 128 + B - 1) / B, B, 0, stream>>>(Wm2, Wm2T, 128, 128);
    pack_bias_kernel<<<2, B, 0, stream>>>(bq, bk, bv, bqkv);
    // 2) q/k/v GEMMs: one 16-row tile per wave
    int mtiles = (N + 15) / 16;
    qkv_kernel<<<(mtiles + 7) / 8, B, 0, stream>>>(hb, WqkvT, bqkv, qkvb, N, mtiles);
    // 3) attention logits + segment max
    int ntiles = (E + 15) / 16;
    int lgrid  = (ntiles + 7) / 8;
    logits_kernel<<<lgrid, B, 0, stream>>>(qb, kbuf, eidx, ea, We1, be1, We2T, be2,
                                           Wa1T, ba1, Wa2, ba2, logits, segmax, E, ntiles);
    // 4) exp + segment sum
    ex_kernel<<<(E + B - 1) / B, B, 0, stream>>>(eidx + E, segmax, logits, segsum, E);
    // 5) messages + weighted scatter
    msg_kernel<<<lgrid, B, 0, stream>>>(vb, eidx, ea, We1, be1, We2T, be2,
                                        Wm1T, bm1, Wm2T, bm2, logits, segsum, out, E, ntiles);
}
